// MultiHeadAttention_31009663877727
// MI455X (gfx1250) — matrix operand order repaired and checked by reading, not compiled, not run on hardware
//
#include <hip/hip_runtime.h>
#include <stdint.h>

typedef __attribute__((ext_vector_type(16))) _Float16 v16h;
typedef __attribute__((ext_vector_type(4)))  _Float16 v4h;
typedef __attribute__((ext_vector_type(8)))  float    v8f;

#define DEVINL __device__ __forceinline__

constexpr int B_ = 2, S_ = 2048, D_ = 1024, H_ = 16, DK_ = 64;

DEVINL v8f wmma_f16(v16h a, v16h b, v8f c) {
  // D = A(16x32 f16) * B(32x16 f16) + C(16x16 f32)
  return __builtin_amdgcn_wmma_f32_16x16x32_f16(
      /*neg_a=*/false, a, /*neg_b=*/false, b,
      /*c_mod=*/(short)0, c, /*reuse_a=*/false, /*reuse_b=*/false);
}

// ---- CDNA5 async global->LDS copy (ASYNCcnt-tracked), via inline asm ----
// VDST VGPR = LDS byte address (low 32 bits of a generic LDS pointer),
// VADDR = 64-bit global address. 16 bytes per lane per issue.
DEVINL void async_ld_b128(uint32_t ldsAddr, const void* gaddr) {
  asm volatile("global_load_async_to_lds_b128 %0, %1, off"
               :: "v"(ldsAddr), "v"(gaddr) : "memory");
}
DEVINL void wait_async0() {
  asm volatile("s_wait_asynccnt 0x0" ::: "memory");
}

// Build a 16-half fragment from two aligned 16-byte runs -> 2x b128 loads.
DEVINL v16h load_frag2(const _Float16* p0, const _Float16* p1) {
  union { v16h h; uint4 u[2]; } f;
  f.u[0] = *reinterpret_cast<const uint4*>(p0);
  f.u[1] = *reinterpret_cast<const uint4*>(p1);
  return f.h;
}

// A 16x32 (MxK) fragment, row-major source with leading dim ld (halves).
DEVINL v16h load_a_frag(const _Float16* base, int row0, int k0, int ld) {
  const int lane = threadIdx.x & 31;
  const int m    = lane & 15;
  const int koff = (lane < 16) ? 0 : 8;
  const _Float16* rp = base + (size_t)(row0 + m) * ld + k0 + koff;
  return load_frag2(rp, rp + 16);
}

// B 32x16 (KxN) fragment from a K-contiguous-per-column source:
// element B[k][n] = base[(n0+n)*col_stride + k0 + k].
DEVINL v16h load_b_cm(const _Float16* base, size_t col_stride, int n0, int k0) {
  const int lane = threadIdx.x & 31;
  const int n    = lane & 15;
  const int kh   = (lane < 16) ? 0 : 8;
  const _Float16* p = base + (size_t)(n0 + n) * col_stride + k0 + kh;
  return load_frag2(p, p + 16);
}

// ---------------- fp32 -> f16 conversion ----------------
__global__ void cvt_f32_to_f16_x4(const float4* __restrict__ src,
                                  v4h* __restrict__ dst, int n4) {
  int i = blockIdx.x * blockDim.x + threadIdx.x;
  if (i < n4) {
    float4 f = src[i];
    v4h h;
    h.x = (_Float16)f.x; h.y = (_Float16)f.y;
    h.z = (_Float16)f.z; h.w = (_Float16)f.w;
    dst[i] = h;
  }
}

// -------- WMMA GEMM: Y[m][n] = sum_k A[m][k]*W[n][k] + bias[n] --------
// MODE 0: store f16 as [b][h][s][dk]  (Q and K layout)
// MODE 1: store f16 as [b][h][dk][s]  (V transposed layout)
// MODE 2: store f32 row-major [m][n]  (final output)
template <int MODE>
__global__ void __launch_bounds__(128)
gemm16(const _Float16* __restrict__ A, const _Float16* __restrict__ W,
       const float* __restrict__ bias, void* __restrict__ dstv, int Ksz, int Nsz) {
  const int wave = threadIdx.x >> 5;
  const int lane = threadIdx.x & 31;
  const int n0 = blockIdx.x * 128 + wave * 32;
  const int m0 = blockIdx.y * 16;

  v8f c0 = {}, c1 = {};
  for (int k0 = 0; k0 < Ksz; k0 += 32) {
    v16h af = load_a_frag(A, m0, k0, Ksz);
    v16h b0 = load_b_cm(W, Ksz, n0, k0);       // W[N][K] is K-contiguous
    v16h b1 = load_b_cm(W, Ksz, n0 + 16, k0);
    c0 = wmma_f16(af, b0, c0);
    c1 = wmma_f16(af, b1, c1);
  }

  const int nloc = lane & 15;
  const int mhi  = (lane < 16) ? 0 : 8;
#pragma unroll
  for (int t = 0; t < 2; ++t) {
    const v8f& c = t ? c1 : c0;
    const int n = n0 + t * 16 + nloc;
    const float bn = bias[n];
#pragma unroll
    for (int v = 0; v < 8; ++v) {
      const int m = m0 + v + mhi;
      const float val = c[v] + bn;
      if (MODE == 2) {
        reinterpret_cast<float*>(dstv)[(size_t)m * Nsz + n] = val;
      } else {
        _Float16* dst = reinterpret_cast<_Float16*>(dstv);
        const int bb = m >> 11;          // m / S_
        const int s  = m & (S_ - 1);
        const int hh = n >> 6;           // n / DK_
        const int dk = n & (DK_ - 1);
        size_t idx;
        if (MODE == 0) idx = (((size_t)(bb * H_ + hh)) * S_ + s) * DK_ + dk;
        else           idx = (((size_t)(bb * H_ + hh)) * DK_ + dk) * S_ + s;
        dst[idx] = (_Float16)val;
      }
    }
  }
}

// ---------------- Flash attention ----------------
// 8 waves/block, all on the SAME (b,h): one wave async-stages the shared
// 32-key K/V tiles to LDS (double buffered), every wave computes its own
// 16-row Q block from the LDS tiles.
__global__ void __launch_bounds__(256)
flash_attn(const _Float16* __restrict__ Q,   // [b][h][s][dk]
           const _Float16* __restrict__ K,   // [b][h][s][dk]
           const _Float16* __restrict__ Vt,  // [b][h][dk][s]
           const int* __restrict__ mask,     // [b][s]
           _Float16* __restrict__ AO) {      // [b][s][D]
  __shared__ __align__(16) _Float16 Pt[8][16 * 32];     // per-wave P staging
  __shared__ __align__(16) _Float16 KT[2][32 * DK_];    // 2 x 4KB key tiles
  __shared__ __align__(16) _Float16 VT[2][DK_ * 32];    // 2 x 4KB value tiles

  const int lane = threadIdx.x & 31;
  const int wv   = threadIdx.x >> 5;
  const int gw   = blockIdx.x * 8 + wv;
  const int qblk = gw & (S_ / 16 - 1);
  const int bh   = gw / (S_ / 16);        // constant across the block
  const int hh   = bh & (H_ - 1);
  const int bb   = bh / H_;
  const int q0   = qblk * 16;

  const _Float16* Qb = Q  + (size_t)bh * S_ * DK_;
  const _Float16* Kb = K  + (size_t)bh * S_ * DK_;
  const _Float16* Vb = Vt + (size_t)bh * DK_ * S_;
  const int* mk = mask + (size_t)bb * S_;

  // Issue one 32-key K/V tile (8KB) into LDS buffer p: 16 async b128/lane.
  auto issue_tile = [&](int s0, int p) {
    // K tile: 32 rows x 128B, lane -> key row `lane`
    {
      const _Float16* g = Kb + (size_t)(s0 + lane) * DK_;
      const uint32_t l = (uint32_t)(uintptr_t)(&KT[p][lane * DK_]);
#pragma unroll
      for (int j = 0; j < 8; ++j) async_ld_b128(l + j * 16, g + j * 8);
    }
    // V tile: 64 rows x 64B, lane -> dk rows `lane`, `lane+32`
#pragma unroll
    for (int r = 0; r < 2; ++r) {
      const int dk = lane + r * 32;
      const _Float16* g = Vb + (size_t)dk * S_ + s0;
      const uint32_t l = (uint32_t)(uintptr_t)(&VT[p][dk * 32]);
#pragma unroll
      for (int j = 0; j < 4; ++j) async_ld_b128(l + j * 16, g + j * 8);
    }
  };

  const v16h aq0 = load_a_frag(Qb, q0, 0, DK_);
  const v16h aq1 = load_a_frag(Qb, q0, 32, DK_);

  float mRow[8], lRow[8];
#pragma unroll
  for (int i = 0; i < 8; ++i) { mRow[i] = -1e30f; lRow[i] = 0.f; }
  v8f acc0 = {}, acc1 = {}, acc2 = {}, acc3 = {};

  _Float16* Pw = &Pt[wv][0];
  const int nloc = lane & 15;
  const int mhi  = (lane < 16) ? 0 : 8;

  if (wv == 0) issue_tile(0, 0);                 // prologue: stage tile 0

  for (int it = 0; it < S_ / 32; ++it) {
    const int s0 = it * 32;
    const int p  = it & 1;
    if (wv == 0) wait_async0();                  // tile `it` resident
    __syncthreads();                             // publish to all waves
    if (wv == 0 && it + 1 < S_ / 32)
      issue_tile(s0 + 32, 1 - p);                // overlap next tile w/ compute

    const _Float16* Kt_ = &KT[p][0];
    const _Float16* Vt_ = &VT[p][0];

    // scores (16 x 32 keys): 2 N-tiles x 2 K-chunks over DK=64 (from LDS)
    v8f sc0 = {}, sc1 = {};
    {
      v16h bk;
      bk = load_b_cm(Kt_, DK_, 0,  0);  sc0 = wmma_f16(aq0, bk, sc0);
      bk = load_b_cm(Kt_, DK_, 0,  32); sc0 = wmma_f16(aq1, bk, sc0);
      bk = load_b_cm(Kt_, DK_, 16, 0);  sc1 = wmma_f16(aq0, bk, sc1);
      bk = load_b_cm(Kt_, DK_, 16, 32); sc1 = wmma_f16(aq1, bk, sc1);
    }
    const bool ok0 = mk[s0 + nloc] != 0;
    const bool ok1 = mk[s0 + 16 + nloc] != 0;

    float corr[8];
#pragma unroll
    for (int v = 0; v < 8; ++v) {
      float x0 = ok0 ? sc0[v] * 0.125f : -1e9f;   // 1/sqrt(64)
      float x1 = ok1 ? sc1[v] * 0.125f : -1e9f;
      float rm = fmaxf(x0, x1);
#pragma unroll
      for (int off = 1; off < 16; off <<= 1)
        rm = fmaxf(rm, __shfl_xor(rm, off, 16));  // row spans a 16-lane group
      const float mnew = fmaxf(mRow[v], rm);
      const float cf = __expf(mRow[v] - mnew);
      const float p0 = __expf(x0 - mnew);
      const float p1 = __expf(x1 - mnew);
      float rs = p0 + p1;
#pragma unroll
      for (int off = 1; off < 16; off <<= 1)
        rs += __shfl_xor(rs, off, 16);
      mRow[v] = mnew;
      lRow[v] = lRow[v] * cf + rs;
      corr[v] = cf;
      const int mIdx = v + mhi;                   // C-layout row of this VGPR
      Pw[mIdx * 32 + nloc]      = (_Float16)p0;
      Pw[mIdx * 32 + 16 + nloc] = (_Float16)p1;
    }
#pragma unroll
    for (int v = 0; v < 8; ++v) {
      acc0[v] *= corr[v]; acc1[v] *= corr[v];
      acc2[v] *= corr[v]; acc3[v] *= corr[v];
    }
    // Re-load P (16x32) in A-fragment layout from LDS: 2x b128 per lane.
    const v16h pa = load_a_frag(Pw, 0, 0, 32);
    // B[k=key][n=dk] from LDS V tile [dk][32]: key-contiguous per dk.
    v16h bv;
    bv = load_b_cm(Vt_, 32, 0,  0); acc0 = wmma_f16(pa, bv, acc0);
    bv = load_b_cm(Vt_, 32, 16, 0); acc1 = wmma_f16(pa, bv, acc1);
    bv = load_b_cm(Vt_, 32, 32, 0); acc2 = wmma_f16(pa, bv, acc2);
    bv = load_b_cm(Vt_, 32, 48, 0); acc3 = wmma_f16(pa, bv, acc3);
  }

#pragma unroll
  for (int v = 0; v < 8; ++v) {
    const float inv = 1.f / lRow[v];
    const int m = q0 + v + mhi;
    const size_t base = ((size_t)bb * S_ + m) * D_ + hh * DK_ + nloc;
    AO[base]      = (_Float16)(acc0[v] * inv);
    AO[base + 16] = (_Float16)(acc1[v] * inv);
    AO[base + 32] = (_Float16)(acc2[v] * inv);
    AO[base + 48] = (_Float16)(acc3[v] * inv);
  }
}

extern "C" void kernel_launch(void* const* d_in, const int* in_sizes, int n_in,
                              void* d_out, int out_size, void* d_ws, size_t ws_size,
                              hipStream_t stream) {
  (void)in_sizes; (void)n_in; (void)out_size; (void)ws_size;
  const float* q    = (const float*)d_in[0];
  const float* k    = (const float*)d_in[1];
  const float* v    = (const float*)d_in[2];
  const int*   mask = (const int*)d_in[3];
  const float* Wq   = (const float*)d_in[4];
  const float* bq   = (const float*)d_in[5];
  const float* Wk   = (const float*)d_in[6];
  const float* bk   = (const float*)d_in[7];
  const float* Wv   = (const float*)d_in[8];
  const float* bv   = (const float*)d_in[9];
  const float* Wo   = (const float*)d_in[10];
  const float* bo   = (const float*)d_in[11];
  float* out = (float*)d_out;

  constexpr size_t BSD = (size_t)B_ * S_ * D_;   // 4,194,304
  constexpr size_t DD  = (size_t)D_ * D_;        // 1,048,576
  _Float16* w   = (_Float16*)d_ws;               // 56 MB total (fits 192MB L2)
  _Float16* qh  = w;
  _Float16* kh  = qh + BSD;
  _Float16* vh  = kh + BSD;
  _Float16* Qh  = vh + BSD;
  _Float16* Kh  = Qh + BSD;
  _Float16* Vt  = Kh + BSD;
  _Float16* Wqh = Vt + BSD;
  _Float16* Wkh = Wqh + DD;
  _Float16* Wvh = Wkh + DD;
  _Float16* Woh = Wvh + DD;
  _Float16* AO  = qh;  // qh dead after the Q projection; reuse for attn out

  auto cvt = [&](const float* s, _Float16* d, size_t n) {
    const int n4 = (int)(n / 4);
    cvt_f32_to_f16_x4<<<(n4 + 255) / 256, 256, 0, stream>>>(
        (const float4*)s, (v4h*)d, n4);
  };
  cvt(q, qh, BSD);  cvt(k, kh, BSD);  cvt(v, vh, BSD);
  cvt(Wq, Wqh, DD); cvt(Wk, Wkh, DD); cvt(Wv, Wvh, DD); cvt(Wo, Woh, DD);

  const int M = B_ * S_;                 // 4096
  dim3 ggrid(D_ / 128, M / 16);          // (8, 256)
  dim3 gblock(128);                      // 4 waves, two 16x16 C-tiles each
  gemm16<0><<<ggrid, gblock, 0, stream>>>(qh, Wqh, bq, (void*)Qh, D_, D_);
  gemm16<0><<<ggrid, gblock, 0, stream>>>(kh, Wkh, bk, (void*)Kh, D_, D_);
  gemm16<1><<<ggrid, gblock, 0, stream>>>(vh, Wvh, bv, (void*)Vt, D_, D_);

  const int nqwaves = B_ * H_ * (S_ / 16);   // 4096 Q blocks
  flash_attn<<<nqwaves / 8, 256, 0, stream>>>(Qh, Kh, Vt, mask, AO);

  gemm16<2><<<ggrid, gblock, 0, stream>>>(AO, Woh, bo, (void*)out, D_, D_);
}
